// SemanticGraphModule_90460601189044
// MI455X (gfx1250) — matrix operand order repaired and checked by reading, not compiled, not run on hardware
//
#include <hip/hip_runtime.h>
#include <hip/hip_bf16.h>
#include <math.h>

// ---------------------------------------------------------------------------
// SemanticGraphModule for MI455X (gfx1250, wave32, WMMA + TDM).
//  * Project the R=5000 unique relation embeddings ONCE, then gather.
//  * softmax over a length-1 key axis == 1.0 -> q/k/Wq/Wk are dead code.
//  * GEMMs: v_wmma_f32_16x16x32_f16, 64x64 block tile, 8 wave32,
//    2 WMMA/wave/K-step; X-side register-pipelined b128 loads + cvt_pk;
//    W-side pre-converted to f16 and DMA'd by TENSOR_LOAD_TO_LDS into a
//    double-buffered LDS tile (TENSORcnt-synchronized), padded by the TDM
//    descriptor to the 80B row stride the WMMA fragment loads expect.
// ---------------------------------------------------------------------------

typedef _Float16 v16h  __attribute__((ext_vector_type(16)));
typedef float    v8f   __attribute__((ext_vector_type(8)));
typedef float    f32x4 __attribute__((ext_vector_type(4)));
typedef unsigned u32x4 __attribute__((ext_vector_type(4)));
typedef unsigned tdm_u4 __attribute__((ext_vector_type(4)));
typedef int      tdm_i4 __attribute__((ext_vector_type(4)));
typedef int      tdm_i8 __attribute__((ext_vector_type(8)));

#define DM    256
#define NNBR  32
#define TOPK  8
#define LDS_STRIDE 40   // halves; 80B row stride -> 16B-aligned b128, TDM pad

#if __has_builtin(__builtin_amdgcn_tensor_load_to_lds)
#define SG_USE_TDM 1
#if __has_include(<hip/amd_detail/amd_gfx1250_TDM.h>)
#define SG_TDM_6ARG 1   // therock/clang-23 headers -> 6-arg builtin
#endif
#else
#define SG_USE_TDM 0
#endif

#if SG_USE_TDM
// Issue a TDM load of a 64(row) x 32(col) f16 tile from W16[O,K] (row-major,
// tile origin = gsrc) into LDS at byte offset ldsOff. D# per ISA 8.3/8.4:
// data_size=1(2B), pad_enable, pad_interval=16 DW (one 64B row),
// pad_amount=4 DW (16B) -> LDS rows land at 80B stride.
__device__ __forceinline__ void sg_tdm_load_w(const _Float16* gsrc,
                                              unsigned ldsOff, int K, int O)
{
    unsigned long long ga = (unsigned long long)(uintptr_t)gsrc;
    tdm_u4 g0;
    g0[0] = 1u;                                   // count=1, user mode
    g0[1] = ldsOff;                               // lds_addr (bytes)
    g0[2] = (unsigned)(ga & 0xFFFFFFFFull);       // global_addr[31:0]
    g0[3] = (unsigned)((ga >> 32) & 0x1FFFFFFull) // global_addr[56:32]
          | (2u << 30);                           // type=2 ("image")
    tdm_i8 g1;
    g1[0] = (int)((1u << 16)                      // data_size = 2 bytes
                | (1u << 20)                      // pad_enable
                | (3u << 22)                      // pad_interval: 16 DWORDs
                | (3u << 25));                    // pad_amount:   4 DWORDs
    g1[1] = (int)(((unsigned)K & 0xFFFFu) << 16); // tensor_dim0[15:0]
    g1[2] = (int)(((unsigned)K >> 16) | (((unsigned)O & 0xFFFFu) << 16));
    g1[3] = (int)(((unsigned)O >> 16) | (32u << 16)); // tile_dim0 = 32
    g1[4] = 64;                                   // tile_dim1 = 64 rows
    g1[5] = K;                                    // tensor_dim0_stride lo32
    g1[6] = 0;
    g1[7] = 0;
    tdm_i4 gz4 = {0, 0, 0, 0};
#if defined(SG_TDM_6ARG)
    tdm_i8 gz8 = {0, 0, 0, 0, 0, 0, 0, 0};
    __builtin_amdgcn_tensor_load_to_lds(g0, g1, gz4, gz4, gz8, 0);
#else
    __builtin_amdgcn_tensor_load_to_lds(g0, g1, gz4, gz4, 0);
#endif
}
#endif

// -------- one-time fp32 -> fp16 weight conversion --------------------------
__global__ __launch_bounds__(256)
void sg_f32tof16_kernel(const float* __restrict__ src,
                        _Float16* __restrict__ dst, int n)
{
    int i = blockIdx.x * blockDim.x + threadIdx.x;
    if (i < n) dst[i] = (_Float16)src[i];
}

// ---------------- GEMM: Y[M,O] = act(X[M,K] @ W16[O,K]^T + bias) ------------
__global__ __launch_bounds__(256)
void sg_gemm_kernel(const float* __restrict__ X, const _Float16* __restrict__ W16,
                    const float* __restrict__ bias, float* __restrict__ Y,
                    int M, int K, int O, int act)
{
    __shared__ _Float16 As[64 * LDS_STRIDE];       // [m][k] f16 (cvt from f32)
    __shared__ _Float16 Bs[2 * 64 * LDS_STRIDE];   // [o][k] f16, double buffer

    const int t    = threadIdx.x;
    const int lane = t & 31;
    const int wave = t >> 5;          // 0..7
    const int hi   = lane >> 4;
    const int lo16 = lane & 15;
    const int mo   = (wave & 3) * 16;
    const int no   = (wave >> 2) * 32;
    const int mBase = blockIdx.y * 64;
    const int oBase = blockIdx.x * 64;

    const int srow = t >> 2;
    const int scol = (t & 3) * 8;
    int gmA = mBase + srow; if (gmA >= M) gmA = M - 1;  // rows >= M never stored
    const int goB = oBase + srow;

    v8f c0 = {0.f,0.f,0.f,0.f,0.f,0.f,0.f,0.f};
    v8f c1 = {0.f,0.f,0.f,0.f,0.f,0.f,0.f,0.f};
    const int nk = K >> 5;

    f32x4 xa0, xa1;                 // X pipeline registers
#if !SG_USE_TDM
    u32x4 wreg;                     // fallback: W f16 pipeline register
#endif

    // ---- prologue: tile 0 in flight ----
    {
        const float* xp = X + (size_t)gmA * K + scol;
        xa0 = *(const f32x4*)xp;  xa1 = *(const f32x4*)(xp + 4);
#if SG_USE_TDM
        if (wave == 0)
            sg_tdm_load_w(W16 + (size_t)oBase * K,
                          (unsigned)(uintptr_t)&Bs[0], K, O);
#else
        wreg = *(const u32x4*)(W16 + (size_t)goB * K + scol);
#endif
    }

    for (int kt = 0; kt < nk; ++kt) {
        const int buf = (kt & 1) * 64 * LDS_STRIDE;
        // stage X regs -> As (fp32 -> fp16, one b128 store)
        {
            union { _Float16 h[8]; u32x4 u; } ha;
            ha.h[0]=(_Float16)xa0[0]; ha.h[1]=(_Float16)xa0[1];
            ha.h[2]=(_Float16)xa0[2]; ha.h[3]=(_Float16)xa0[3];
            ha.h[4]=(_Float16)xa1[0]; ha.h[5]=(_Float16)xa1[1];
            ha.h[6]=(_Float16)xa1[2]; ha.h[7]=(_Float16)xa1[3];
            *(u32x4*)&As[srow * LDS_STRIDE + scol] = ha.u;
        }
#if SG_USE_TDM
        if (wave == 0) __builtin_amdgcn_s_wait_tensorcnt(0); // tile kt landed
#else
        *(u32x4*)&Bs[buf + srow * LDS_STRIDE + scol] = wreg;
#endif
        __syncthreads();

        // next tile in flight while WMMAs below run
        if (kt + 1 < nk) {
            const int kb = (kt + 1) << 5;
            const float* xp = X + (size_t)gmA * K + kb + scol;
            xa0 = *(const f32x4*)xp;  xa1 = *(const f32x4*)(xp + 4);
#if SG_USE_TDM
            if (wave == 0)
                sg_tdm_load_w(W16 + (size_t)oBase * K + kb,
                              (unsigned)(uintptr_t)&Bs[buf ^ (64 * LDS_STRIDE)],
                              K, O);
#else
            wreg = *(const u32x4*)(W16 + (size_t)goB * K + kb + scol);
#endif
        }
        if (kt + 2 < nk && t < 64) {            // X prefetch two tiles ahead
            int gm = mBase + t; if (gm >= M) gm = M - 1;
            __builtin_prefetch(X + (size_t)gm * K + ((kt + 2) << 5), 0, 1);
        }

        // fragment loads per CDNA5 16-bit A(16x32)/B(32x16) VGPR layouts
        union { v16h v; unsigned u[8]; } fa, fb0, fb1;
        #pragma unroll
        for (int vv = 0; vv < 8; ++vv) {
            const int ka  = ((vv & 4) ? 16 : 0) + hi * 8 + (vv & 3) * 2;
            fa.u[vv]  = *(const unsigned*)&As[(mo + lo16) * LDS_STRIDE + ka];
            const int kbb = ((vv & 4) ? 16 : 0) + hi * 8 + (vv & 3) * 2;
            fb0.u[vv] = *(const unsigned*)&Bs[buf + (no + lo16) * LDS_STRIDE + kbb];
            fb1.u[vv] = *(const unsigned*)&Bs[buf + (no + 16 + lo16) * LDS_STRIDE + kbb];
        }
        c0 = __builtin_amdgcn_wmma_f32_16x16x32_f16(false, fa.v, false, fb0.v,
                                                    (short)0, c0, false, false);
        c1 = __builtin_amdgcn_wmma_f32_16x16x32_f16(false, fa.v, false, fb1.v,
                                                    (short)0, c1, false, false);
        __syncthreads();
    }

    // epilogue: C VGPR i -> M = hi*8 + i, N = lo16
    const int n0 = oBase + no + lo16;
    const int n1 = n0 + 16;
    const float bn0 = bias[n0], bn1 = bias[n1];
    #pragma unroll
    for (int i = 0; i < 8; ++i) {
        const int m = mBase + mo + hi * 8 + i;
        if (m < M) {
            float y0 = c0[i] + bn0;
            float y1 = c1[i] + bn1;
            if (act == 1) { // exact GELU
                y0 = 0.5f * y0 * (1.0f + erff(y0 * 0.70710678118654752f));
                y1 = 0.5f * y1 * (1.0f + erff(y1 * 0.70710678118654752f));
            }
            Y[(size_t)m * O + n0] = y0;
            Y[(size_t)m * O + n1] = y1;
        }
    }
}

// -------- inverse L2 norms of P rows (for cosine similarity) ---------------
__global__ __launch_bounds__(32)
void sg_rownorm_kernel(const float* __restrict__ P, float* __restrict__ Pinv)
{
    int r = blockIdx.x, lane = threadIdx.x;
    const float* p = P + (size_t)r * DM + lane * 8;
    float ss = 0.f;
    #pragma unroll
    for (int j = 0; j < 8; ++j) ss += p[j] * p[j];
    for (int off = 16; off >= 1; off >>= 1) ss += __shfl_xor(ss, off, 32);
    if (lane == 0) Pinv[r] = 1.0f / fmaxf(sqrtf(ss), 1e-12f);
}

// -------- per-edge: cosine sims, top-8 (stable), ctx mean, states init -----
__global__ __launch_bounds__(32)
void sg_topk_ctx_kernel(const int* __restrict__ rel_ids,
                        const int* __restrict__ nbr_ids,
                        const float* __restrict__ P,
                        const float* __restrict__ Pinv,
                        float* __restrict__ ctx, float* __restrict__ states)
{
    int e = blockIdx.x, lane = threadIdx.x;
    int rid = rel_ids[e];
    int nid = nbr_ids[(size_t)e * NNBR + lane];

    const float* pr = P + (size_t)rid * DM;
    const float* pn = P + (size_t)nid * DM;
    float acc = 0.f;
    #pragma unroll 8
    for (int d = 0; d < DM; ++d) acc += pn[d] * pr[d];
    float sim = acc * Pinv[nid] * Pinv[rid];

    unsigned selmask = 0;
    float cacc[8] = {0.f,0.f,0.f,0.f,0.f,0.f,0.f,0.f};

    for (int ts = 0; ts < TOPK; ++ts) {
        float v  = ((selmask >> lane) & 1u) ? -INFINITY : sim;
        int  idx = lane;
        for (int off = 16; off >= 1; off >>= 1) {  // argmax, ties -> lowest lane
            float ov = __shfl_xor(v, off, 32);
            int   oi = __shfl_xor(idx, off, 32);
            if (ov > v || (ov == v && oi < idx)) { v = ov; idx = oi; }
        }
        selmask |= 1u << idx;
        int snid = __shfl(nid, idx, 32);
        const float* ps = P + (size_t)snid * DM + lane * 8;
        #pragma unroll
        for (int j = 0; j < 8; ++j) cacc[j] += ps[j];
    }

    float* co = ctx    + (size_t)e * DM + lane * 8;
    float* so = states + (size_t)e * DM + lane * 8;
    const float* pr8 = pr + lane * 8;
    #pragma unroll
    for (int j = 0; j < 8; ++j) { co[j] = cacc[j] * 0.125f; so[j] = pr8[j]; }
}

// -------- fused residual + LayerNorm (+ optional edge mask) ----------------
__global__ __launch_bounds__(256)
void sg_ln_kernel(const float* __restrict__ resid, const float* __restrict__ delta,
                  const float* __restrict__ g, const float* __restrict__ b,
                  const float* __restrict__ mask, float* __restrict__ out)
{
    __shared__ float red[8];
    int row = blockIdx.x, t = threadIdx.x, lane = t & 31, w = t >> 5;
    float x = resid[(size_t)row * DM + t] + delta[(size_t)row * DM + t];

    float s = x;
    for (int off = 16; off >= 1; off >>= 1) s += __shfl_xor(s, off, 32);
    if (lane == 0) red[w] = s;
    __syncthreads();
    float tot = 0.f;
    #pragma unroll
    for (int i = 0; i < 8; ++i) tot += red[i];
    float mu = tot * (1.0f / 256.0f);
    __syncthreads();

    float d = x - mu, s2 = d * d;
    for (int off = 16; off >= 1; off >>= 1) s2 += __shfl_xor(s2, off, 32);
    if (lane == 0) red[w] = s2;
    __syncthreads();
    float tot2 = 0.f;
    #pragma unroll
    for (int i = 0; i < 8; ++i) tot2 += red[i];
    float var = tot2 * (1.0f / 256.0f);

    float y = d * rsqrtf(var + 1e-5f) * g[t] + b[t];
    if (mask) y *= mask[row];
    out[(size_t)row * DM + t] = y;
}

// -------- append edge_mask to d_out tail -----------------------------------
__global__ void sg_tail_kernel(const float* __restrict__ mask,
                               float* __restrict__ out_tail, int n)
{
    int i = blockIdx.x * blockDim.x + threadIdx.x;
    if (i < n) out_tail[i] = mask[i];
}

extern "C" void kernel_launch(void* const* d_in, const int* in_sizes, int n_in,
                              void* d_out, int out_size, void* d_ws, size_t ws_size,
                              hipStream_t stream)
{
    const int*   rel_ids   = (const int*)d_in[0];
    const int*   nbr       = (const int*)d_in[1];
    const float* edge_mask = (const float*)d_in[2];
    const float* rel_emb   = (const float*)d_in[3];
    const float* proj_W    = (const float*)d_in[4];
    const float* proj_b    = (const float*)d_in[5];
    // d_in[6..9] = Wq,bq,Wk,bk: dead code (softmax over a single key == 1).
    const float* Wv   = (const float*)d_in[10];
    const float* bv   = (const float*)d_in[11];
    const float* Wo   = (const float*)d_in[12];
    const float* bo   = (const float*)d_in[13];
    const float* ln1g = (const float*)d_in[14];
    const float* ln1b = (const float*)d_in[15];
    const float* ln2g = (const float*)d_in[16];
    const float* ln2b = (const float*)d_in[17];
    const float* W1   = (const float*)d_in[18];
    const float* b1   = (const float*)d_in[19];
    const float* W2   = (const float*)d_in[20];
    const float* b2   = (const float*)d_in[21];
    float* out = (float*)d_out;

    const int REL = 512, Dff = 1024, L = 2;
    const int Rn  = in_sizes[3] / REL;  // 5000 unique relations
    const int BE  = in_sizes[0];        // 4096 edges

    // workspace layout
    float* ws     = (float*)d_ws;
    float* P      = ws;                                     // [Rn, 256]
    float* Pinv   = P + (size_t)Rn * DM;                    // [Rn]
    float* ctx    = Pinv + (size_t)((Rn + 255) & ~255);     // [BE, 256]
    float* states = ctx + (size_t)BE * DM;                  // [BE, 256]
    float* buf1   = states + (size_t)BE * DM;               // [BE, 1024]
    float* buf2   = buf1 + (size_t)BE * Dff;                // [BE, 256]
    _Float16* h16 = (_Float16*)(buf2 + (size_t)BE * DM);    // f16 weight blob
    _Float16* projW16 = h16;                                // [256,512]
    _Float16* Wv16    = projW16 + (size_t)DM * REL;         // [L,256,256]
    _Float16* Wo16    = Wv16 + (size_t)L * DM * DM;
    _Float16* W116    = Wo16 + (size_t)L * DM * DM;         // [L,1024,256]
    _Float16* W216    = W116 + (size_t)L * Dff * DM;        // [L,256,1024]

    auto cvt = [&](const float* s, _Float16* d, int n) {
        sg_f32tof16_kernel<<<(n + 255) / 256, 256, 0, stream>>>(s, d, n);
    };
    cvt(proj_W, projW16, DM * REL);
    cvt(Wv, Wv16, L * DM * DM);
    cvt(Wo, Wo16, L * DM * DM);
    cvt(W1, W116, L * Dff * DM);
    cvt(W2, W216, L * DM * Dff);

    auto gemm = [&](const float* X, const _Float16* W16, const float* bias,
                    float* Y, int M, int K, int O, int act) {
        dim3 g(O / 64, (M + 63) / 64);
        sg_gemm_kernel<<<g, 256, 0, stream>>>(X, W16, bias, Y, M, K, O, act);
    };

    // 1) project all unique relation embeddings once
    gemm(rel_emb, projW16, proj_b, P, Rn, REL, DM, 0);
    sg_rownorm_kernel<<<Rn, 32, 0, stream>>>(P, Pinv);

    // 2) cosine top-8 neighbor context + initial states via gather
    sg_topk_ctx_kernel<<<BE, 32, 0, stream>>>(rel_ids, nbr, P, Pinv, ctx, states);

    // 3) transformer layers (attention collapses to Wo(Wv ctx + bv) + bo)
    for (int l = 0; l < L; ++l) {
        gemm(ctx,  Wv16 + (size_t)l * DM * DM,  bv + l * DM,  buf2, BE, DM,  DM,  0);
        gemm(buf2, Wo16 + (size_t)l * DM * DM,  bo + l * DM,  buf1, BE, DM,  DM,  0);
        sg_ln_kernel<<<BE, 256, 0, stream>>>(states, buf1, ln1g + l * DM,
                                             ln1b + l * DM, nullptr, states);
        gemm(states, W116 + (size_t)l * Dff * DM, b1 + l * Dff, buf1, BE, DM,  Dff, 1);
        gemm(buf1,   W216 + (size_t)l * DM * Dff, b2 + l * DM,  buf2, BE, Dff, DM,  0);
        float* lnout = (l == L - 1) ? out : states;
        sg_ln_kernel<<<BE, 256, 0, stream>>>(states, buf2, ln2g + l * DM,
                                             ln2b + l * DM, edge_mask, lnout);
    }

    // 4) second output of the tuple: edge_mask
    sg_tail_kernel<<<(BE + 255) / 256, 256, 0, stream>>>(edge_mask,
                                                         out + (size_t)BE * DM, BE);
}
